// TargetForecast_model_56126632624348
// MI455X (gfx1250) — matrix operand order repaired and checked by reading, not compiled, not run on hardware
//
#include <hip/hip_runtime.h>
#include <hip/hip_bf16.h>

typedef __attribute__((ext_vector_type(16))) _Float16 v16h;
typedef __attribute__((ext_vector_type(8)))  _Float16 v8h;
typedef __attribute__((ext_vector_type(4)))  _Float16 v4h;
typedef __attribute__((ext_vector_type(8)))  float    v8f;
typedef __attribute__((ext_vector_type(4)))  float    v4f;

#define EPSF 1e-8f

namespace {
constexpr int D      = 128;
constexpr int BU     = 64;
constexpr int L      = 512;
constexpr int P0     = 200000;
constexpr int P1     = 50000;
constexpr int NEG_MS = 4;
constexpr int NCOLS  = BU * L;          // 32768 total sequence positions

// workspace layout (bytes)
constexpr size_t OFF_ACC = 0;                                    // 8 floats
constexpr size_t OFF_W   = 256;                                  // 4 * D*D f16 (Wq,Wk,Wv,Wh)
constexpr size_t SZ_W    = (size_t)4 * D * D * 2;                // 131072
constexpr size_t OFF_IT  = OFF_W + SZ_W;                         // 131328 (256-aligned)
constexpr size_t SZ_MAT  = (size_t)NCOLS * D * 2;                // 8 MB (f16 L*B x D)
constexpr size_t OFF_QT  = OFF_IT + SZ_MAT;
constexpr size_t OFF_KT  = OFF_QT + SZ_MAT;
constexpr size_t OFF_V   = OFF_KT + SZ_MAT;                      // row-major per user: [u][d][t]
constexpr size_t OFF_BT  = OFF_V + SZ_MAT;
constexpr size_t OFF_OT  = OFF_BT + SZ_MAT;
} // namespace

// ---------- helpers ----------

__device__ __forceinline__ float waveReduceSum(float x) {
    x += __shfl_xor(x, 16, 32);
    x += __shfl_xor(x, 8, 32);
    x += __shfl_xor(x, 4, 32);
    x += __shfl_xor(x, 2, 32);
    x += __shfl_xor(x, 1, 32);
    return x;
}

__device__ __forceinline__ float neg_logsig(float x) {
    float s = 1.0f / (1.0f + __expf(-x));
    return -__logf(s + EPSF);
}

__device__ __forceinline__ v8f wmma16(v16h a, v16h b, v8f c) {
    // v_wmma_f32_16x16x32_f16
    return __builtin_amdgcn_wmma_f32_16x16x32_f16(false, a, false, b, (short)0, c, false, false);
}

// A operand: 16x32 f16 tile from row-major matrix (ld elements per row).
// lane 0-15: M=lane, K=0..15; lane 16-31: M=lane-16, K=16..31
__device__ __forceinline__ v16h loadA(const _Float16* __restrict__ p, int ld,
                                      int m0, int k0, int lane) {
    int row = m0 + (lane & 15);
    int klo = (lane >> 4) * 16;
    const _Float16* base = p + (size_t)row * ld + k0;
    union { v16h v; v8h h[2]; } u;
    u.h[0] = *(const v8h*)(base + klo);
    u.h[1] = *(const v8h*)(base + 8 + klo);
    return u.v;
}

// B operand: 32x16 f16 tile from "K contiguous per column" storage (N x ld).
// lane 0-15: N=lane, K=0..15; lane 16-31: N=lane-16, K=16..31
__device__ __forceinline__ v16h loadB(const _Float16* __restrict__ p, int ld,
                                      int n0, int k0, int lane) {
    int col = n0 + (lane & 15);
    int ko  = k0 + (lane >> 4) * 16;
    return *(const v16h*)(p + (size_t)col * ld + ko);
}

// ---------- init: zero accumulators, convert 4 weight mats f32->f16 ----------

__global__ void init_kernel(const float* __restrict__ Wq, const float* __restrict__ Wk,
                            const float* __restrict__ Wv, const float* __restrict__ Wh,
                            _Float16* __restrict__ W16, float* __restrict__ acc) {
    int i = blockIdx.x * blockDim.x + threadIdx.x;
    if (i < 8) acc[i] = 0.0f;
    const int M = D * D; // 16384, power of two
    for (int idx = i; idx < 4 * M; idx += gridDim.x * blockDim.x) {
        const float* src = (idx < M) ? Wq : (idx < 2 * M) ? Wk : (idx < 3 * M) ? Wv : Wh;
        W16[idx] = (_Float16)src[idx & (M - 1)];
    }
}

// ---------- build I^T (f16):  IT[(u*L+t)*D + d] = emb[q[u,t]][d] + W_rep[d, c[u,t]] ----------

__global__ void build_I_kernel(const float* __restrict__ emb, const float* __restrict__ Wrep,
                               const int* __restrict__ qs, const int* __restrict__ cs,
                               _Float16* __restrict__ IT) {
    int lane = threadIdx.x & 31;
    int wid  = (blockIdx.x * blockDim.x + threadIdx.x) >> 5;
    if (wid >= NCOLS) return;
    int qi = qs[wid];
    int c  = cs[wid];
    int d  = lane * 4;
    v4f e = *(const v4f*)(emb + (size_t)qi * D + d);
    v4h o;
    for (int r = 0; r < 4; ++r)
        o[r] = (_Float16)(e[r] + Wrep[(d + r) * 2 + c]);
    *(v4h*)(IT + (size_t)wid * D + d) = o;
}

// ---------- batched GEMM: Out = W(128x128) @ Bm(128 x NCOLS) ----------
// One wave per 16-column strip: computes all 8 M-tiles, so each B tile is
// loaded once and reused for 8 WMMAs (B traffic / 8 vs tile-per-wave).
// All 8 A tiles of a k-step are batch-loaded before the WMMA chain so the
// loads clause together and waits are progressive instead of full drains.
// transposed=1: Out stored (NCOLS x D) column-t-major-over-d (B-operand layout)
// transposed=0: Out stored per-user row-major [u][d][t] (A-operand layout)

__global__ void gemm128_kernel(const _Float16* __restrict__ W, const _Float16* __restrict__ Bm,
                               _Float16* __restrict__ Out, int transposed) {
    int lane = threadIdx.x & 31;
    int wid  = (blockIdx.x * blockDim.x + threadIdx.x) >> 5;
    const int nstrips = NCOLS / 16; // 2048
    if (wid >= nstrips) return;
    int n0 = wid * 16;
    v8f acc[8] = {};
#pragma unroll
    for (int k0 = 0; k0 < D; k0 += 32) {
        v16h b = loadB(Bm, D, n0, k0, lane);
        v16h a[8];
#pragma unroll
        for (int m = 0; m < 8; ++m) a[m] = loadA(W, D, m * 16, k0, lane);
#pragma unroll
        for (int m = 0; m < 8; ++m) acc[m] = wmma16(a[m], b, acc[m]);
    }
    int half = lane >> 4;
    if (transposed) {
        int col = n0 + (lane & 15);
#pragma unroll
        for (int m = 0; m < 8; ++m) {
            int mb = m * 16 + half * 8;
            v8h h;
            for (int r = 0; r < 8; ++r) h[r] = (_Float16)acc[m][r];
            *(v8h*)(Out + (size_t)col * D + mb) = h;
        }
    } else {
        int n = n0 + (lane & 15);
        int u = n >> 9;
        int t = n & (L - 1);
        _Float16* dst = Out + (size_t)u * D * L + t;
#pragma unroll
        for (int m = 0; m < 8; ++m) {
            int mb = m * 16 + half * 8;
            for (int r = 0; r < 8; ++r)
                dst[(size_t)(mb + r) * L] = (_Float16)acc[m][r];
        }
    }
}

// ---------- causal flash attention, one wave per 16-query tile ----------
// b[:,j] = sum_{i<=j} v[:,i] * softmax_i(k[:,i].q[:,j] * scale); output bT (L x D per user)

__global__ void attn_kernel(const _Float16* __restrict__ qT, const _Float16* __restrict__ kT,
                            const _Float16* __restrict__ vR, _Float16* __restrict__ bT) {
    __shared__ __align__(32) _Float16 lds[8][32 * 16]; // per-wave 32 keys x 16 cols
    int lane = threadIdx.x & 31;
    int warp = threadIdx.x >> 5;
    int wid  = blockIdx.x * 8 + warp;
    if (wid >= BU * (L / 16)) return;
    int u  = wid >> 5;            // 32 query tiles per user
    int j0 = (wid & 31) * 16;
    const _Float16* q = qT + (size_t)u * L * D;
    const _Float16* k = kT + (size_t)u * L * D;
    const _Float16* v = vR + (size_t)u * D * L;
    const float scale = 0.088388347648318447f; // 1/sqrt(128)

    int col  = lane & 15;
    int half = lane >> 4;
    int jcol = j0 + col;

    // Q tile is invariant across the key loop: keep all 4 k-step B-operands
    // in registers (32 VGPRs) instead of re-streaming from global.
    v16h qb[4];
#pragma unroll
    for (int kk = 0; kk < 4; ++kk) qb[kk] = loadB(q, D, j0, kk * 32, lane);

    float mrun = -INFINITY, lrun = 0.0f;
    v8f acc[8] = {};

    _Float16* slab = &lds[warp][0];
    int nkb = ((j0 + 15) >> 5) + 1;
    for (int kb = 0; kb < nkb; ++kb) {
        int i0 = kb * 32;
        // S = K^T @ Q for 32 keys x 16 queries (two 16x16 D tiles)
        v8f s0 = {}, s1 = {};
#pragma unroll
        for (int kk = 0; kk < 4; ++kk) {
            int k0 = kk * 32;
            v16h a0 = loadA(k, D, i0, k0, lane);
            v16h a1 = loadA(k, D, i0 + 16, k0, lane);
            s0 = wmma16(a0, qb[kk], s0);
            s1 = wmma16(a1, qb[kk], s1);
        }
        // scale + causal mask (key index must be <= query index), local max
        float mloc = -INFINITY;
        for (int r = 0; r < 8; ++r) {
            int key0 = i0 + half * 8 + r;
            int key1 = key0 + 16;
            float x0 = (key0 <= jcol) ? s0[r] * scale : -INFINITY;
            float x1 = (key1 <= jcol) ? s1[r] * scale : -INFINITY;
            s0[r] = x0; s1[r] = x1;
            mloc = fmaxf(mloc, fmaxf(x0, x1));
        }
        mloc = fmaxf(mloc, __shfl_xor(mloc, 16, 32));   // full column max
        float mnew = fmaxf(mrun, mloc);
        float corr = __expf(mrun - mnew);
        float psum = 0.0f;
        for (int r = 0; r < 8; ++r) {
            float p0 = __expf(s0[r] - mnew);
            float p1 = __expf(s1[r] - mnew);
            s0[r] = p0; s1[r] = p1;
            psum += p0 + p1;
        }
        psum += __shfl_xor(psum, 16, 32);
        lrun = lrun * corr + psum;
        mrun = mnew;
        for (int dt = 0; dt < 8; ++dt)
            for (int r = 0; r < 8; ++r) acc[dt][r] *= corr;

        // reshape P (D-layout) -> B-operand layout through LDS
        v8h p0h, p1h;
        for (int r = 0; r < 8; ++r) { p0h[r] = (_Float16)s0[r]; p1h[r] = (_Float16)s1[r]; }
        *(v8h*)(slab + col * 32 + half * 8)      = p0h;
        *(v8h*)(slab + col * 32 + 16 + half * 8) = p1h;
        asm volatile("s_wait_dscnt 0x0" ::: "memory");  // cross-lane LDS visibility in-wave
        v16h bp = *(const v16h*)(slab + col * 32 + half * 16);

        // acc_d += V(16d x 32keys) @ P(32keys x 16q)
#pragma unroll
        for (int dt = 0; dt < 8; ++dt) {
            v16h a = loadA(v, L, dt * 16, i0, lane);
            acc[dt] = wmma16(a, bp, acc[dt]);
        }
    }
    float rinv = 1.0f / lrun;
#pragma unroll
    for (int dt = 0; dt < 8; ++dt) {
        v8h h;
        for (int r = 0; r < 8; ++r) h[r] = (_Float16)(acc[dt][r] * rinv);
        *(v8h*)(bT + ((size_t)u * L + j0 + col) * D + dt * 16 + half * 8) = h;
    }
}

// ---------- edge task 0: -(sum logsig(pos) + sum logsig(-neg)) ----------

__global__ void edge0_kernel(const float* __restrict__ emb, const int* __restrict__ pos,
                             const int* __restrict__ neg, float* __restrict__ acc) {
    int lane = threadIdx.x & 31;
    int wid  = (blockIdx.x * blockDim.x + threadIdx.x) >> 5;
    int nw   = (gridDim.x * blockDim.x) >> 5;
    float local = 0.0f;
    for (int idx = wid; idx < 2 * P0; idx += nw) {
        const int* pr = (idx < P0) ? (pos + 2 * idx) : (neg + 2 * (idx - P0));
        float sgn = (idx < P0) ? 1.0f : -1.0f;
        int nidx = idx + nw;
        if (nidx < 2 * P0) {
            const int* npr = (nidx < P0) ? (pos + 2 * nidx) : (neg + 2 * (nidx - P0));
            __builtin_prefetch(npr, 0, 3);
        }
        int h = pr[0], t = pr[1];
        v4f a = *(const v4f*)(emb + (size_t)h * D + lane * 4);
        v4f b = *(const v4f*)(emb + (size_t)t * D + lane * 4);
        float d = a[0] * b[0] + a[1] * b[1] + a[2] * b[2] + a[3] * b[3];
        d = waveReduceSum(d);
        local += neg_logsig(sgn * d);
    }
    if (lane == 0) atomicAdd(acc, local);
}

// ---------- edge task 1: margin-style with tiled positives ----------

__global__ void edge1_kernel(const float* __restrict__ emb, const int* __restrict__ pos,
                             const int* __restrict__ neg, float* __restrict__ acc) {
    int lane = threadIdx.x & 31;
    int wid  = (blockIdx.x * blockDim.x + threadIdx.x) >> 5;
    int nw   = (gridDim.x * blockDim.x) >> 5;
    const int total = P1 * NEG_MS;
    float local = 0.0f;
    for (int idx = wid; idx < total; idx += nw) {
        int j = idx % P1;
        int ph = pos[2 * j], pt = pos[2 * j + 1];
        int nh = neg[2 * idx], nt = neg[2 * idx + 1];
        int d = lane * 4;
        v4f eph = *(const v4f*)(emb + (size_t)ph * D + d);
        v4f ept = *(const v4f*)(emb + (size_t)pt * D + d);
        v4f enh = *(const v4f*)(emb + (size_t)nh * D + d);
        v4f ent = *(const v4f*)(emb + (size_t)nt * D + d);
        float dp = 0.f, dh = 0.f, dt2 = 0.f;
        for (int r = 0; r < 4; ++r) {
            dp  += eph[r] * ept[r];
            dh  += enh[r] * ept[r];
            dt2 += eph[r] * ent[r];
        }
        dp  = waveReduceSum(dp);
        dh  = waveReduceSum(dh);
        dt2 = waveReduceSum(dt2);
        local += neg_logsig(dp - dh) + neg_logsig(dp - dt2);
    }
    if (lane == 0) atomicAdd(acc + 1, local);
}

// ---------- sequence BCE with shifted user state ----------

__global__ void seq_bce_kernel(const _Float16* __restrict__ oT, const float* __restrict__ uinit,
                               const float* __restrict__ emb, const int* __restrict__ qs,
                               const int* __restrict__ cs, float* __restrict__ acc) {
    int lane = threadIdx.x & 31;
    int wid  = (blockIdx.x * blockDim.x + threadIdx.x) >> 5;
    if (wid >= NCOLS) return;
    int t = wid & (L - 1);
    int d = lane * 4;
    v4f uq;
    if (t == 0) {
        uq = *(const v4f*)(uinit + d);
    } else {
        v4h hh = *(const v4h*)(oT + (size_t)(wid - 1) * D + d); // u*L + (t-1)
        uq[0] = (float)hh[0]; uq[1] = (float)hh[1];
        uq[2] = (float)hh[2]; uq[3] = (float)hh[3];
    }
    int qi = qs[wid];
    v4f e = *(const v4f*)(emb + (size_t)qi * D + d);
    float dot = uq[0] * e[0] + uq[1] * e[1] + uq[2] * e[2] + uq[3] * e[3];
    dot = waveReduceSum(dot);
    float pred = 1.0f / (1.0f + __expf(-dot));
    float cf = (float)cs[wid];
    float contrib = -(cf * __logf(pred + EPSF) + (1.0f - cf) * __logf(1.0f - pred + EPSF));
    if (lane == 0) atomicAdd(acc + 2, contrib);
}

// ---------- finalize: task-uncertainty weighting ----------

__global__ void finalize_kernel(const float* __restrict__ acc, const float* __restrict__ sd0p,
                                const float* __restrict__ sd1p, const float* __restrict__ sdsp,
                                float* __restrict__ out) {
    if (threadIdx.x == 0 && blockIdx.x == 0) {
        float sd0 = *sd0p, sd1 = *sd1p, sds = *sdsp;
        float tf = acc[0] / (sd0 * sd0 + EPSF) + 400000.0f * __logf(sd0 + EPSF)
                 + acc[1] / (sd1 * sd1 + EPSF) + 400000.0f * __logf(sd1 + EPSF)
                 + acc[2] / (sds * sds + EPSF) + 512.0f   * __logf(sds + EPSF);
        out[0] = tf;
    }
}

// ---------- host launcher ----------

extern "C" void kernel_launch(void* const* d_in, const int* in_sizes, int n_in,
                              void* d_out, int out_size, void* d_ws, size_t ws_size,
                              hipStream_t stream) {
    (void)in_sizes; (void)n_in; (void)out_size; (void)ws_size;
    const float* emb    = (const float*)d_in[0];
    const float* W_rep  = (const float*)d_in[1];
    const float* W_q    = (const float*)d_in[2];
    const float* W_k    = (const float*)d_in[3];
    const float* W_v    = (const float*)d_in[4];
    const float* Headag = (const float*)d_in[5];
    const float* uinit  = (const float*)d_in[6];
    const float* sd_e0  = (const float*)d_in[7];
    const float* sd_e1  = (const float*)d_in[8];
    const float* sd_seq = (const float*)d_in[9];
    const int* pos_e0   = (const int*)d_in[10];
    const int* neg_e0   = (const int*)d_in[11];
    const int* pos_e1   = (const int*)d_in[12];
    const int* neg_e1   = (const int*)d_in[13];
    const int* q_seqs   = (const int*)d_in[14];
    const int* c_seqs   = (const int*)d_in[15];
    float* out = (float*)d_out;

    char* ws = (char*)d_ws;
    float*    acc = (float*)(ws + OFF_ACC);
    _Float16* W16 = (_Float16*)(ws + OFF_W);
    _Float16* IT  = (_Float16*)(ws + OFF_IT);
    _Float16* qT  = (_Float16*)(ws + OFF_QT);
    _Float16* kT  = (_Float16*)(ws + OFF_KT);
    _Float16* vR  = (_Float16*)(ws + OFF_V);
    _Float16* bT  = (_Float16*)(ws + OFF_BT);
    _Float16* oT  = (_Float16*)(ws + OFF_OT);

    _Float16* Wq16 = W16;
    _Float16* Wk16 = W16 + (size_t)D * D;
    _Float16* Wv16 = W16 + (size_t)2 * D * D;
    _Float16* Wh16 = W16 + (size_t)3 * D * D;

    // 1) zero accumulators + weight f32->f16
    init_kernel<<<64, 256, 0, stream>>>(W_q, W_k, W_v, Headag, W16, acc);

    // 2) build I^T (f16) for all users
    build_I_kernel<<<NCOLS / 8, 256, 0, stream>>>(emb, W_rep, q_seqs, c_seqs, IT);

    // 3) q/k/v GEMMs (WMMA), one wave per 16-column strip (2048 waves)
    const int gemm_blocks = (NCOLS / 16) / 8; // 256
    gemm128_kernel<<<gemm_blocks, 256, 0, stream>>>(Wq16, IT, qT, 1);
    gemm128_kernel<<<gemm_blocks, 256, 0, stream>>>(Wk16, IT, kT, 1);
    gemm128_kernel<<<gemm_blocks, 256, 0, stream>>>(Wv16, IT, vR, 0);

    // 4) causal flash attention (WMMA + LDS reshape)
    attn_kernel<<<BU * (L / 16) / 8, 256, 0, stream>>>(qT, kT, vR, bT);

    // 5) o = Head_agg @ b  (WMMA), stored transposed (L x D)
    gemm128_kernel<<<gemm_blocks, 256, 0, stream>>>(Wh16, bT, oT, 1);

    // 6) sequence BCE with shifted user state
    seq_bce_kernel<<<NCOLS / 8, 256, 0, stream>>>(oT, uinit, emb, q_seqs, c_seqs, acc);

    // 7) edge losses (bandwidth-bound gathers)
    edge0_kernel<<<2048, 256, 0, stream>>>(emb, pos_e0, neg_e0, acc);
    edge1_kernel<<<2048, 256, 0, stream>>>(emb, pos_e1, neg_e1, acc);

    // 8) combine with task uncertainty terms
    finalize_kernel<<<1, 32, 0, stream>>>(acc, sd_e0, sd_e1, sd_seq, out);
}
